// MultiHeadAttention_68367289418154
// MI455X (gfx1250) — matrix operand order repaired and checked by reading, not compiled, not run on hardware
//
#include <hip/hip_runtime.h>
#include <hip/hip_bf16.h>

// ---------------- problem constants ----------------
#define EMBED   768
#define HEADS   12
#define HEADDIM 64
#define BATCH   2
#define SEQ     2048
#define NTOK    (BATCH * SEQ)          // 4096
#define NBH     (BATCH * HEADS)        // 24
#define SCALE   0.125f                 // 1/sqrt(64)

typedef _Float16 v16h __attribute__((ext_vector_type(16)));
typedef _Float16 v8h  __attribute__((ext_vector_type(8)));
typedef _Float16 v4h  __attribute__((ext_vector_type(4)));
typedef float    v8f  __attribute__((ext_vector_type(8)));
typedef float    v4f  __attribute__((ext_vector_type(4)));

// LDS layout for the attention kernel (bytes)
#define LDS_SC    0        // 16 x 2048 fp32 scores (128 KB); f16 weights packed
                           // in-place into bytes [r*8192, r*8192+4096) per row r
#define LDS_PC    131072   // 4 x 16x16 fp32 partial-C tiles for split-K (4 KB)
#define LDS_Q     135168   // 16 x 64 f16 Q tile staged via async copy (2 KB)
#define LDS_TOTAL 137216   // 2 blocks / WGP (274432 <= 327680)

// ---------------- WMMA fragment helpers ----------------
// A-matrix 16x32 f16 (row-major source, leading dim ld halves):
// lane L: row = L&15, K-chunks [8*hi, 8*hi+8) and [16+8*hi, 16+8*hi+8), hi=L>>4
template <typename P>
static __device__ __forceinline__ v16h load_a_f16(P base, int ld, int lane) {
    const int r  = lane & 15;
    const int ko = (lane >> 4) * 8;
    v8h c0 = *(const v8h*)(base + r * ld + ko);
    v8h c1 = *(const v8h*)(base + r * ld + 16 + ko);
    return __builtin_shufflevector(c0, c1, 0,1,2,3,4,5,6,7,8,9,10,11,12,13,14,15);
}

static __device__ __forceinline__ v8f wmma_f16(v16h a, v16h b, v8f c) {
    return __builtin_amdgcn_wmma_f32_16x16x32_f16(false, a, false, b, (short)0, c, false, false);
}

// ---------------- kernel 0: fp32 -> fp16 convert (vec4) ----------------
__global__ void cvt_f32_to_f16_kernel(const float* __restrict__ src,
                                      _Float16* __restrict__ dst, int n4) {
    int i = blockIdx.x * blockDim.x + threadIdx.x;
    if (i < n4) {
        v4f s = ((const v4f*)src)[i];
        v4h h;
#pragma unroll
        for (int j = 0; j < 4; ++j) h[j] = (_Float16)s[j];
        ((v4h*)dst)[i] = h;
    }
}

// ---------------- kernel 1: fused QKV projection ----------------
//   Q -> Qh  [b,h,s,d]   (A operand of scores GEMM)
//   K -> KhT [b,h,d,s]   (B operand of scores GEMM, pre-transposed)
//   V -> Vh  [b,h,s,d]   (B operand of attn@V GEMM)
__global__ void proj_qkv_kernel(const _Float16* __restrict__ xh,
                                const _Float16* __restrict__ wq,
                                const _Float16* __restrict__ wk,
                                const _Float16* __restrict__ wv,
                                const float* __restrict__ bq,
                                const float* __restrict__ bk,
                                const float* __restrict__ bv,
                                _Float16* __restrict__ Qh,
                                _Float16* __restrict__ KhT,
                                _Float16* __restrict__ Vh) {
    const int lane = threadIdx.x & 31;
    const int wave = threadIdx.x >> 5;
    const int NT_M = NTOK / 16;   // 256
    const int NT_N = EMBED / 16;  // 48
    long tile = (long)blockIdx.x * 8 + wave;     // 3 * 256 * 48 tiles total
    const int qkv = (int)(tile / (NT_M * NT_N));
    int rem = (int)(tile % (NT_M * NT_N));
    const int m0 = (rem / NT_N) * 16;
    const int n0 = (rem % NT_N) * 16;

    const _Float16* W  = (qkv == 0) ? wq : (qkv == 1) ? wk : wv;
    const float*    bi = (qkv == 0) ? bq : (qkv == 1) ? bk : bv;

    v8f c = {};
#pragma unroll 4
    for (int kt = 0; kt < EMBED / 32; ++kt) {
        const int k0 = kt * 32;
        __builtin_prefetch(W + (k0 + 32 + lane) * EMBED + n0, 0, 1);
        v16h a = load_a_f16(xh + (long)m0 * EMBED + k0, EMBED, lane);
        v16h b;
#pragma unroll
        for (int s = 0; s < 16; ++s)
            b[s] = W[(long)(k0 + ((s & 8) << 1) + 8 * (lane >> 4) + (s & 7)) * EMBED + n0 + (lane & 15)];
        c = wmma_f16(a, b, c);
    }

    const int n  = n0 + (lane & 15);
    const int hi = lane >> 4;
    const int h  = n >> 6;           // n / 64
    const int d  = n & 63;
    const float bval = bi[n];
#pragma unroll
    for (int j = 0; j < 8; ++j) {
        const int m = m0 + 8 * hi + j;
        const int b_ = m >> 11;      // m / 2048
        const int s  = m & 2047;
        const int bh = b_ * HEADS + h;
        const _Float16 val = (_Float16)(c[j] + bval);
        if (qkv == 0)      Qh [((long)bh * SEQ + s) * HEADDIM + d] = val;
        else if (qkv == 1) KhT[((long)bh * HEADDIM + d) * SEQ + s] = val;
        else               Vh [((long)bh * SEQ + s) * HEADDIM + d] = val;
    }
}

// ---------------- kernel 2: fused attention ----------------
// One block per (b,h, 16-query block). Full 16x2048 fp32 score block lives in
// LDS; the 403 MB attn_weights output touches HBM exactly once (write-only).
__global__ void attention_kernel(const _Float16* __restrict__ Qh,
                                 const _Float16* __restrict__ KhT,
                                 const _Float16* __restrict__ Vh,
                                 float* __restrict__ attnW,
                                 _Float16* __restrict__ attnH) {
    extern __shared__ char smem[];
    float* sc = (float*)(smem + LDS_SC);
    const int lane = threadIdx.x & 31;
    const int wave = threadIdx.x >> 5;
    const int qb = blockIdx.x & 127;              // 128 query blocks
    const int bh = blockIdx.x >> 7;               // 24 (b,h) pairs
    const int b_ = bh / HEADS;
    const int h  = bh % HEADS;
    const int q0 = qb * 16;

    // ---- stage Q tile (16x64 f16 = 2 KB) into LDS via async copy ----
    const _Float16* qbase = Qh + ((long)bh * SEQ + q0) * HEADDIM;
    if (threadIdx.x < 128) {
        unsigned lds_off = (unsigned)(LDS_Q + threadIdx.x * 16);
        const void* g = (const void*)((const char*)qbase + threadIdx.x * 16);
        asm volatile("global_load_async_to_lds_b128 %0, %1, off"
                     :: "v"(lds_off), "v"(g) : "memory");
    }
    asm volatile("s_wait_asynccnt 0x0" ::: "memory");
    __syncthreads();

    // ---- phase 1: scores = (Q @ K^T) * scale ----
    const _Float16* qlds = (const _Float16*)(smem + LDS_Q);
    v16h a0 = load_a_f16(qlds,      HEADDIM, lane);   // d = 0..31
    v16h a1 = load_a_f16(qlds + 32, HEADDIM, lane);   // d = 32..63
    const _Float16* kbase = KhT + (long)bh * HEADDIM * SEQ;

    for (int t = wave; t < SEQ / 16; t += 8) {
        const int k0 = t * 16;
        v16h b0, b1;
#pragma unroll
        for (int s = 0; s < 16; ++s) {
            const int kd = ((s & 8) << 1) + 8 * (lane >> 4) + (s & 7);
            b0[s] = kbase[(long)kd * SEQ + k0 + (lane & 15)];
            b1[s] = kbase[(long)(32 + kd) * SEQ + k0 + (lane & 15)];
        }
        v8f c = {};
        c = wmma_f16(a0, b0, c);
        c = wmma_f16(a1, b1, c);
        const int n  = lane & 15;
        const int hi = lane >> 4;
#pragma unroll
        for (int j = 0; j < 8; ++j)
            sc[(8 * hi + j) * SEQ + k0 + n] = c[j] * SCALE;
    }
    __syncthreads();

    // ---- phase 2: softmax (each wave owns 2 rows; vec4 everywhere) ----
#pragma unroll
    for (int rr = 0; rr < 2; ++rr) {
        const int row = 2 * wave + rr;
        v4f* srow4 = (v4f*)(sc + row * SEQ);              // 512 v4f per row
        v4f mx4 = {-3.402823466e38f, -3.402823466e38f, -3.402823466e38f, -3.402823466e38f};
        for (int i = lane; i < SEQ / 4; i += 32) {
            v4f t = srow4[i];
#pragma unroll
            for (int j = 0; j < 4; ++j) mx4[j] = fmaxf(mx4[j], t[j]);
        }
        float mx = fmaxf(fmaxf(mx4[0], mx4[1]), fmaxf(mx4[2], mx4[3]));
#pragma unroll
        for (int off = 16; off >= 1; off >>= 1) mx = fmaxf(mx, __shfl_xor(mx, off, 32));

        v4f s4 = {};
        for (int i = lane; i < SEQ / 4; i += 32) {
            v4f t = srow4[i];
#pragma unroll
            for (int j = 0; j < 4; ++j) t[j] = __expf(t[j] - mx);
            srow4[i] = t;
            s4 += t;
        }
        float sum = s4[0] + s4[1] + s4[2] + s4[3];
#pragma unroll
        for (int off = 16; off >= 1; off >>= 1) sum += __shfl_xor(sum, off, 32);
        const float inv = 1.f / sum;

        // normalize: write fp32 weights to HBM (write-once) and pack an f16
        // copy IN PLACE over the first half of this row's own LDS storage.
        v4f* wrow4 = (v4f*)(attnW + ((long)bh * SEQ + q0 + row) * SEQ);
        v4h* hrow4 = (v4h*)(smem + LDS_SC + row * (SEQ * 4));
        for (int i = lane; i < SEQ / 4; i += 32) {
            v4f t = srow4[i] * inv;   // LDS read precedes in-place f16 write
            wrow4[i] = t;             // coalesced global_store_b128
            v4h hh;
#pragma unroll
            for (int j = 0; j < 4; ++j) hh[j] = (_Float16)t[j];
            hrow4[i] = hh;
        }
    }
    __syncthreads();

    // ---- phase 3: attn = P @ V, split-K over 8 waves ----
    // f16 weight row r lives at half-offset r*4096 (ld = 4096 halves).
    const _Float16* wlds = (const _Float16*)(smem + LDS_SC);
    float* pc = (float*)(smem + LDS_PC);
    {
        const int n0    = (wave & 3) * 16;
        const int khalf = wave >> 2;
        v8f c = {};
        const _Float16* vbase = Vh + (long)bh * SEQ * HEADDIM + n0;
#pragma unroll 4
        for (int kt = khalf * 32; kt < khalf * 32 + 32; ++kt) {
            const int k0 = kt * 32;
            v16h a = load_a_f16(wlds + k0, 4096, lane);   // ds_load_b128 x2
            v16h b;
#pragma unroll
            for (int s = 0; s < 16; ++s)
                b[s] = vbase[(long)(k0 + ((s & 8) << 1) + 8 * (lane >> 4) + (s & 7)) * HEADDIM + (lane & 15)];
            c = wmma_f16(a, b, c);
        }
        const int n  = lane & 15;
        const int hi = lane >> 4;
        if (wave >= 4) {
#pragma unroll
            for (int j = 0; j < 8; ++j)
                pc[(wave - 4) * 256 + (8 * hi + j) * 16 + n] = c[j];
        }
        __syncthreads();
        if (wave < 4) {
            const int e = h * HEADDIM + n0 + n;
#pragma unroll
            for (int j = 0; j < 8; ++j) {
                const float v = c[j] + pc[wave * 256 + (8 * hi + j) * 16 + n];
                const long m = (long)b_ * SEQ + q0 + 8 * hi + j;
                attnH[m * EMBED + e] = (_Float16)v;
            }
        }
    }
}

// ---------------- kernel 3: output projection ----------------
__global__ void proj_out_kernel(const _Float16* __restrict__ ah,
                                const _Float16* __restrict__ wo,
                                const float* __restrict__ bo,
                                float* __restrict__ out) {
    const int lane = threadIdx.x & 31;
    const int wave = threadIdx.x >> 5;
    const int NT_N = EMBED / 16;  // 48
    long tile = (long)blockIdx.x * 8 + wave;
    const int m0 = (int)(tile / NT_N) * 16;
    const int n0 = (int)(tile % NT_N) * 16;

    v8f c = {};
#pragma unroll 4
    for (int kt = 0; kt < EMBED / 32; ++kt) {
        const int k0 = kt * 32;
        __builtin_prefetch(wo + (k0 + 32 + lane) * EMBED + n0, 0, 1);
        v16h a = load_a_f16(ah + (long)m0 * EMBED + k0, EMBED, lane);
        v16h b;
#pragma unroll
        for (int s = 0; s < 16; ++s)
            b[s] = wo[(long)(k0 + ((s & 8) << 1) + 8 * (lane >> 4) + (s & 7)) * EMBED + n0 + (lane & 15)];
        c = wmma_f16(a, b, c);
    }
    const int n  = n0 + (lane & 15);
    const int hi = lane >> 4;
    const float bval = bo[n];
#pragma unroll
    for (int j = 0; j < 8; ++j) {
        const int m = m0 + 8 * hi + j;
        out[(long)m * EMBED + n] = c[j] + bval;
    }
}

// ---------------- workspace layout (bytes) ----------------
#define WS_XH    0
#define WS_WQH   (WS_XH   + (size_t)NTOK * EMBED * 2)
#define WS_WKH   (WS_WQH  + (size_t)EMBED * EMBED * 2)
#define WS_WVH   (WS_WKH  + (size_t)EMBED * EMBED * 2)
#define WS_WOH   (WS_WVH  + (size_t)EMBED * EMBED * 2)
#define WS_QH    (WS_WOH  + (size_t)EMBED * EMBED * 2)
#define WS_KHT   (WS_QH   + (size_t)NBH * SEQ * HEADDIM * 2)
#define WS_VH    (WS_KHT  + (size_t)NBH * SEQ * HEADDIM * 2)
#define WS_AH    (WS_VH   + (size_t)NBH * SEQ * HEADDIM * 2)

extern "C" void kernel_launch(void* const* d_in, const int* in_sizes, int n_in,
                              void* d_out, int out_size, void* d_ws, size_t ws_size,
                              hipStream_t stream) {
    const float* x  = (const float*)d_in[0];
    const float* Wq = (const float*)d_in[1];
    const float* bq = (const float*)d_in[2];
    const float* Wk = (const float*)d_in[3];
    const float* bk = (const float*)d_in[4];
    const float* Wv = (const float*)d_in[5];
    const float* bv = (const float*)d_in[6];
    const float* Wo = (const float*)d_in[7];
    const float* bo = (const float*)d_in[8];

    char* ws = (char*)d_ws;
    _Float16* xh  = (_Float16*)(ws + WS_XH);
    _Float16* wqh = (_Float16*)(ws + WS_WQH);
    _Float16* wkh = (_Float16*)(ws + WS_WKH);
    _Float16* wvh = (_Float16*)(ws + WS_WVH);
    _Float16* woh = (_Float16*)(ws + WS_WOH);
    _Float16* Qh  = (_Float16*)(ws + WS_QH);
    _Float16* KhT = (_Float16*)(ws + WS_KHT);
    _Float16* Vh  = (_Float16*)(ws + WS_VH);
    _Float16* ah  = (_Float16*)(ws + WS_AH);

    float* out   = (float*)d_out;                        // [B,S,E]
    float* attnW = (float*)d_out + (size_t)NTOK * EMBED; // [B,H,S,S]

    // 0) fp32 -> fp16 conversions (vec4)
    const int NX4 = NTOK * EMBED / 4, NW4 = EMBED * EMBED / 4;
    cvt_f32_to_f16_kernel<<<(NX4 + 255) / 256, 256, 0, stream>>>(x,  xh,  NX4);
    cvt_f32_to_f16_kernel<<<(NW4 + 255) / 256, 256, 0, stream>>>(Wq, wqh, NW4);
    cvt_f32_to_f16_kernel<<<(NW4 + 255) / 256, 256, 0, stream>>>(Wk, wkh, NW4);
    cvt_f32_to_f16_kernel<<<(NW4 + 255) / 256, 256, 0, stream>>>(Wv, wvh, NW4);
    cvt_f32_to_f16_kernel<<<(NW4 + 255) / 256, 256, 0, stream>>>(Wo, woh, NW4);

    // 1) fused QKV projection
    proj_qkv_kernel<<<(3 * 256 * 48) / 8, 256, 0, stream>>>(
        xh, wqh, wkh, wvh, bq, bk, bv, Qh, KhT, Vh);

    // 2) fused attention (137,216 B dynamic LDS -> 2 blocks per WGP)
    attention_kernel<<<NBH * (SEQ / 16), 256, LDS_TOTAL, stream>>>(
        Qh, KhT, Vh, attnW, ah);

    // 3) output projection
    proj_out_kernel<<<(256 * 48) / 8, 256, 0, stream>>>(ah, woh, bo, out);
}
